// AWGBGAE_4526895530153
// MI455X (gfx1250) — matrix operand order repaired and checked by reading, not compiled, not run on hardware
//
#include <hip/hip_runtime.h>
#include <hip/hip_bf16.h>

// ---------------------------------------------------------------------------
// CDNA5 (gfx1250) wave32 WMMA bf16 pipeline for:
//   h   = relu(Lap @ (X @ W1))
//   emb = Lap @ (h @ W2)
//   out = softmax(-(pairwise sq dist of emb)) + 1e-10
//
// MI455X roofline design:
//  * Laplacian demoted to bf16: 128MB -> second full pass is L2-resident
//    (192MB global L2); fp32 originals are read with NT loads so they don't
//    evict the bf16 working set.
//  * All GEMM B operands stored column-major so every fragment load is a
//    16-byte vector load (contiguous k order, same for A and B).
//  * Double-buffered k-loops: next k-step fragments load while current WMMAs
//    execute (avoids s_wait_loadcnt 0x0 before every WMMA).
//  * Softmax needs no max pass (dist>=0, dist[i,i]==0). Row sums computed in
//    a store-free pass; output recomputed and written once with NT stores.
// ---------------------------------------------------------------------------

typedef __attribute__((ext_vector_type(16))) __bf16 v16bf;
typedef __attribute__((ext_vector_type(8)))  float  v8f;
typedef __attribute__((ext_vector_type(4)))  float  f32x4;   // NT-load legal

union AF {
    v16bf v;
    unsigned short u[16];
    uint4 q[2];
};

union HalfTile {            // 8 bf16 = one 16B store
    uint4 q;
    unsigned short u[8];
};

__device__ inline v8f zero8() { v8f z = {}; return z; }

__device__ inline unsigned short f2bf(float f) {
    unsigned int u = __float_as_uint(f);
    unsigned int r = u + 0x7FFFu + ((u >> 16) & 1u);   // round-to-nearest-even
    return (unsigned short)(r >> 16);
}

// ---------------------------------------------------------------------------
// fp32 -> bf16 elementwise convert; NT loads (source is dead after this)
// ---------------------------------------------------------------------------
__global__ __launch_bounds__(256) void cvt_f32_bf16_k(
    const float* __restrict__ in, unsigned short* __restrict__ out, size_t n4)
{
    size_t i = (size_t)blockIdx.x * blockDim.x + threadIdx.x;
    if (i >= n4) return;
    const f32x4* p = (const f32x4*)(in + i * 4);
    f32x4 v = __builtin_nontemporal_load(p);
    ushort4 o;
    o.x = f2bf(v.x); o.y = f2bf(v.y); o.z = f2bf(v.z); o.w = f2bf(v.w);
    *(ushort4*)(out + i * 4) = o;
}

// fp32 [R,C] row-major -> bf16 [C,R] row-major (transpose; small matrices)
__global__ __launch_bounds__(256) void cvt_t_f32_bf16_k(
    const float* __restrict__ in, unsigned short* __restrict__ out, int R, int C)
{
    int idx = blockIdx.x * 256 + threadIdx.x;
    if (idx >= R * C) return;
    int r = idx / C, c = idx - r * C;
    out[(size_t)c * R + r] = f2bf(in[idx]);
}

// ---------------------------------------------------------------------------
// bf16 GEMM, f32 accumulate via v_wmma_f32_16x16x32_bf16.
//   A  : row-major [M, lda]  (M dim via grid.y, 128 rows / block, 8 waves)
//   BT : row-major [N, ldbt] == B column-major  -> vector fragment loads
//   C  : CT_STORE ? bf16 transposed [N, ldc] (contiguous b128 lane stores)
//                 : bf16 row-major [M, ldc] (+ optional f32 row-major)
// Each wave: 16 rows x WNT 16-col tiles; k-loop double-buffered. K >= 64.
// ---------------------------------------------------------------------------
template <int WNT, bool RELU, bool CT_STORE, bool STORE_F32>
__global__ __launch_bounds__(256) void gemm_bf16_k(
    const unsigned short* __restrict__ A, const unsigned short* __restrict__ BT,
    unsigned short* __restrict__ Cb, float* __restrict__ Cf,
    int K, int lda, int ldbt, int ldc)
{
    const int wave = threadIdx.x >> 5;
    const int lane = threadIdx.x & 31;
    const int r    = lane & 15;
    const int hi   = lane >> 4;

    const int row0 = blockIdx.y * 128 + wave * 16;
    const int col0 = blockIdx.x * (16 * WNT);

    // A fragment (16x32 MxK), k order matching B:
    // lanes 0-15: K 0..15 ; lanes 16-31: K 16..31
    const unsigned short* arow = A + (size_t)(row0 + r) * lda + hi * 16;
    // B fragment: lanes 0-15 hold K kb..kb+15, lanes 16-31 kb+16..kb+31:
    // column n per lane -> 16 consecutive bf16 of BT row n.
    const unsigned short* brow[WNT];
#pragma unroll
    for (int t = 0; t < WNT; ++t)
        brow[t] = BT + (size_t)(col0 + t * 16 + r) * ldbt + hi * 16;

    v8f acc[WNT];
#pragma unroll
    for (int t = 0; t < WNT; ++t) acc[t] = zero8();

    AF a_cur, b_cur[WNT];
    a_cur.q[0] = *(const uint4*)(arow);
    a_cur.q[1] = *(const uint4*)(arow + 8);
#pragma unroll
    for (int t = 0; t < WNT; ++t) {
        b_cur[t].q[0] = *(const uint4*)(brow[t]);
        b_cur[t].q[1] = *(const uint4*)(brow[t] + 8);
    }

    int kb = 0;
    for (; kb < K - 32; kb += 32) {
        const int kn = kb + 32;
        AF a_nxt, b_nxt[WNT];
        a_nxt.q[0] = *(const uint4*)(arow + kn);
        a_nxt.q[1] = *(const uint4*)(arow + kn + 8);
#pragma unroll
        for (int t = 0; t < WNT; ++t) {
            b_nxt[t].q[0] = *(const uint4*)(brow[t] + kn);
            b_nxt[t].q[1] = *(const uint4*)(brow[t] + kn + 8);
        }
#pragma unroll
        for (int t = 0; t < WNT; ++t)
            acc[t] = __builtin_amdgcn_wmma_f32_16x16x32_bf16(
                false, a_cur.v, false, b_cur[t].v, (short)0, acc[t], false, false);
        a_cur = a_nxt;
#pragma unroll
        for (int t = 0; t < WNT; ++t) b_cur[t] = b_nxt[t];
    }
#pragma unroll
    for (int t = 0; t < WNT; ++t)
        acc[t] = __builtin_amdgcn_wmma_f32_16x16x32_bf16(
            false, a_cur.v, false, b_cur[t].v, (short)0, acc[t], false, false);

#pragma unroll
    for (int t = 0; t < WNT; ++t) {
        const int n = col0 + t * 16 + r;
        if (CT_STORE) {
            HalfTile o;                    // lane's 8 elems are M-contiguous
#pragma unroll
            for (int i = 0; i < 8; ++i) {
                float v = acc[t][i];
                if (RELU) v = fmaxf(v, 0.0f);
                o.u[i] = f2bf(v);
            }
            *(uint4*)(Cb + (size_t)n * ldc + row0 + 8 * hi) = o.q;
        } else {
#pragma unroll
            for (int i = 0; i < 8; ++i) {
                const int m = row0 + i + 8 * hi;
                float v = acc[t][i];
                if (RELU) v = fmaxf(v, 0.0f);
                Cb[(size_t)m * ldc + n] = f2bf(v);
                if (STORE_F32) Cf[(size_t)m * ldc + n] = v;
            }
        }
    }
}

// ---------------------------------------------------------------------------
// sq[i] = sum_k emb[i][k]^2   (emb is [8192,64] f32); one wave per row
// ---------------------------------------------------------------------------
__global__ __launch_bounds__(256) void rowsq_k(
    const float* __restrict__ emb, float* __restrict__ sq)
{
    const int wave = threadIdx.x >> 5;
    const int lane = threadIdx.x & 31;
    const int row  = blockIdx.x * 8 + wave;
    const float* e = emb + (size_t)row * 64;
    float a = e[lane], b = e[lane + 32];
    float s = a * a + b * b;
#pragma unroll
    for (int m = 16; m >= 1; m >>= 1) s += __shfl_xor(s, m, 32);
    if (lane == 0) sq[row] = s;
}

// ---------------------------------------------------------------------------
// Distance/softmax over G = emb @ emb^T via WMMA.
//   Pass 1 (WRITE_OUT=false): S[i] = sum_j exp(-dist(i,j)); store-free.
//   Pass 2 (WRITE_OUT=true):  recompute, write out = exp*invS + 1e-10 once
//     (NT stores; output is never re-read).
// Block: 512 threads = 16 waves; block owns 16 rows x all 8192 cols.
// emb row-major doubles as B^T (B = emb^T). Column loop double-buffered.
// ---------------------------------------------------------------------------
template <bool WRITE_OUT>
__global__ __launch_bounds__(512) void dist_k(
    const unsigned short* __restrict__ Eb,  // emb bf16 [8192,64]
    const float* __restrict__ sq,           // [8192]
    const float* __restrict__ Sin,          // pass2: row sums
    float* __restrict__ out,                // pass2: [8192,8192]
    float* __restrict__ Sout)               // pass1: row sums
{
    __shared__ float rowsum[16];
    const int tid  = threadIdx.x;
    const int wave = tid >> 5;
    const int lane = tid & 31;
    const int r    = lane & 15;
    const int hi   = lane >> 4;
    if (!WRITE_OUT) {
        if (tid < 16) rowsum[tid] = 0.0f;
        __syncthreads();
    }

    const int row0 = blockIdx.x * 16;

    // A fragments: emb rows row0..row0+15, K = 0..63 (two 16x32 frags)
    const unsigned short* arow = Eb + (size_t)(row0 + r) * 64;
    AF a0, a1;
    a0.q[0] = *(const uint4*)(arow +  0 + hi * 16);
    a0.q[1] = *(const uint4*)(arow +  8 + hi * 16);
    a1.q[0] = *(const uint4*)(arow + 32 + hi * 16);
    a1.q[1] = *(const uint4*)(arow + 40 + hi * 16);

    float srow[8], invRow[8];
#pragma unroll
    for (int i = 0; i < 8; ++i) {
        srow[i] = sq[row0 + i + 8 * hi];
        if (WRITE_OUT) invRow[i] = 1.0f / Sin[row0 + i + 8 * hi];
    }

    float lacc[8];
#pragma unroll
    for (int i = 0; i < 8; ++i) lacc[i] = 0.0f;

    const int colstep = wave * 16 + r;      // this lane's column within a tile
    const unsigned short* ebase = Eb + (size_t)colstep * 64 + hi * 16;

    AF b0c, b1c;                             // double-buffered B fragments
    b0c.q[0] = *(const uint4*)(ebase);
    b0c.q[1] = *(const uint4*)(ebase + 8);
    b1c.q[0] = *(const uint4*)(ebase + 32);
    b1c.q[1] = *(const uint4*)(ebase + 40);

    for (int ci = 0; ci < 8192 / 256; ++ci) {
        AF b0n, b1n;
        if (ci + 1 < 8192 / 256) {
            const unsigned short* en = ebase + (size_t)(ci + 1) * 256 * 64;
            b0n.q[0] = *(const uint4*)(en);
            b0n.q[1] = *(const uint4*)(en + 8);
            b1n.q[0] = *(const uint4*)(en + 32);
            b1n.q[1] = *(const uint4*)(en + 40);
        }

        v8f acc = zero8();
        acc = __builtin_amdgcn_wmma_f32_16x16x32_bf16(
            false, a0.v, false, b0c.v, (short)0, acc, false, false);
        acc = __builtin_amdgcn_wmma_f32_16x16x32_bf16(
            false, a1.v, false, b1c.v, (short)0, acc, false, false);

        const int   n   = ci * 256 + colstep;
        const float sqn = sq[n];
#pragma unroll
        for (int i = 0; i < 8; ++i) {
            float d = srow[i] + sqn - 2.0f * acc[i];
            d = fmaxf(d, 0.0f);
            float e = __expf(-d);
            if (WRITE_OUT) {
                float* p = out + (size_t)(row0 + i + 8 * hi) * 8192 + n;
                __builtin_nontemporal_store(e * invRow[i] + 1e-10f, p);
            } else {
                lacc[i] += e;
            }
        }
        b0c = b0n;
        b1c = b1n;
    }

    if (!WRITE_OUT) {
        // reduce within each 16-lane half (masks <16 stay inside the half)
#pragma unroll
        for (int m = 8; m >= 1; m >>= 1) {
#pragma unroll
            for (int i = 0; i < 8; ++i) lacc[i] += __shfl_xor(lacc[i], m, 32);
        }
        if (r == 0) {           // lane 0 (rows 0-7) and lane 16 (rows 8-15)
#pragma unroll
            for (int i = 0; i < 8; ++i) atomicAdd(&rowsum[i + 8 * hi], lacc[i]);
        }
        __syncthreads();
        if (tid < 16) Sout[row0 + tid] = rowsum[tid];
    }
}

// ---------------------------------------------------------------------------
extern "C" void kernel_launch(void* const* d_in, const int* in_sizes, int n_in,
                              void* d_out, int out_size, void* d_ws, size_t ws_size,
                              hipStream_t stream) {
    (void)in_sizes; (void)n_in; (void)out_size; (void)ws_size;
    constexpr int N = 8192, D_IN = 1024, D_MID = 256, D_EMB = 64;

    const float* Lap = (const float*)d_in[0];
    const float* X   = (const float*)d_in[1];
    const float* W1  = (const float*)d_in[2];
    const float* W2  = (const float*)d_in[3];
    float* out = (float*)d_out;

    // workspace carve-up (all offsets well aligned)
    char* ws = (char*)d_ws;
    unsigned short* LapB = (unsigned short*)ws; ws += (size_t)N * N    * 2; // 128MB
    unsigned short* XB   = (unsigned short*)ws; ws += (size_t)N * D_IN * 2; //  16MB
    unsigned short* W1T  = (unsigned short*)ws; ws += (size_t)D_MID * D_IN  * 2; // [256,1024]
    unsigned short* W2T  = (unsigned short*)ws; ws += (size_t)D_EMB * D_MID * 2; // [64,256]
    unsigned short* T1T  = (unsigned short*)ws; ws += (size_t)D_MID * N * 2;     // [256,8192]
    unsigned short* HB   = (unsigned short*)ws; ws += (size_t)N * D_MID * 2;     // [8192,256]
    unsigned short* T2T  = (unsigned short*)ws; ws += (size_t)D_EMB * N * 2;     // [64,8192]
    float*          EmbF = (float*)ws;          ws += (size_t)N * D_EMB * 4;
    unsigned short* EmbB = (unsigned short*)ws; ws += (size_t)N * D_EMB * 2;
    float*          Sq   = (float*)ws;          ws += (size_t)N * 4;
    float*          Ssum = (float*)ws;          ws += (size_t)N * 4;

    // 1) demote inputs to bf16 (weights transposed so they serve as B^T)
    cvt_f32_bf16_k<<<((size_t)N * N    / 4 + 255) / 256, 256, 0, stream>>>(Lap, LapB, (size_t)N * N / 4);
    cvt_f32_bf16_k<<<((size_t)N * D_IN / 4 + 255) / 256, 256, 0, stream>>>(X,   XB,   (size_t)N * D_IN / 4);
    cvt_t_f32_bf16_k<<<(D_IN * D_MID + 255) / 256, 256, 0, stream>>>(W1, W1T, D_IN, D_MID);
    cvt_t_f32_bf16_k<<<(D_MID * D_EMB + 255) / 256, 256, 0, stream>>>(W2, W2T, D_MID, D_EMB);

    // 2) T1 = X @ W1, stored transposed [256,8192] (next use is as B)
    gemm_bf16_k<4, false, true, false><<<dim3(D_MID / 64, N / 128), 256, 0, stream>>>(
        XB, W1T, T1T, nullptr, D_IN, D_IN, D_IN, N);

    // 3) h = relu(Lap @ T1), row-major [8192,256] (next use is as A)
    gemm_bf16_k<4, true, false, false><<<dim3(D_MID / 64, N / 128), 256, 0, stream>>>(
        LapB, T1T, HB, nullptr, N, N, N, D_MID);

    // 4) T2 = h @ W2, stored transposed [64,8192] (next use is as B)
    gemm_bf16_k<2, false, true, false><<<dim3(D_EMB / 32, N / 128), 256, 0, stream>>>(
        HB, W2T, T2T, nullptr, D_MID, D_MID, D_MID, N);

    // 5) emb = Lap @ T2, row-major bf16+f32 (used as A and as B^T; Lap hits L2)
    gemm_bf16_k<2, false, false, true><<<dim3(D_EMB / 32, N / 128), 256, 0, stream>>>(
        LapB, T2T, EmbB, EmbF, N, N, N, D_EMB);

    // 6) sq[i] = |emb_i|^2
    rowsq_k<<<N / 8, 256, 0, stream>>>(EmbF, Sq);

    // 7) pass 1: row sums of exp(-dist) (no output traffic)
    dist_k<false><<<N / 16, 512, 0, stream>>>(EmbB, Sq, nullptr, nullptr, Ssum);

    // 8) pass 2: recompute + single normalized NT write of the 256MB output
    dist_k<true><<<N / 16, 512, 0, stream>>>(EmbB, Sq, Ssum, out, nullptr);
}
